// MoeLayer_66340064854695
// MI455X (gfx1250) — matrix operand order repaired and checked by reading, not compiled, not run on hardware
//
#include <hip/hip_runtime.h>
#include <hip/hip_bf16.h>
#include <stdint.h>

// Problem constants (from reference): B=2,S=1024 -> T=2048 tokens
#define T_TOK 2048
#define HDIM  1024
#define FDIM  2048
#define NEXP  8
#define TOPK  2

typedef __attribute__((ext_vector_type(16))) __bf16       v16bf;
typedef __attribute__((ext_vector_type(8)))  float        v8f;
typedef __attribute__((ext_vector_type(4)))  unsigned int u32x4;
typedef __attribute__((ext_vector_type(8)))  int          i32x8;
typedef __attribute__((ext_vector_type(4)))  int          i32x4;

#if defined(__HIP_DEVICE_COMPILE__) && __has_builtin(__builtin_amdgcn_tensor_load_to_lds) && \
    __has_builtin(__builtin_amdgcn_s_wait_tensorcnt)
#define USE_TDM 1
#else
#define USE_TDM 0
#endif

// ---- WMMA fragment builders (CDNA5 16-bit layouts, CDNA5 ISA §7.12.2) ----
// A 16x32 bf16: lane L -> row M=L%16; lanes 0-15 hold K = kk+{0..7,16..23},
// lanes 16-31 hold K = kk+{8..15,24..31}. arow points at As[m] (bf16, contiguous K).
static __device__ __forceinline__ v16bf frag_a_lds(const __bf16* __restrict__ arow,
                                                   int kk, int lane) {
  const int kbase = (lane < 16) ? 0 : 8;
  const __bf16* p = arow + kk + kbase;
  v16bf a;
#pragma unroll
  for (int j = 0; j < 8; ++j) a[j] = p[j];
#pragma unroll
  for (int j = 0; j < 8; ++j) a[8 + j] = p[16 + j];
  return a;
}

// B 32x16 bf16 from an fp32 LDS tile [Kc][ldn]: lane L -> col n; lanes 0-15 hold
// K = kk+{0..7,16..23}, lanes 16-31 hold K = kk+{8..15,24..31} (cvt f32->bf16 in VALU).
static __device__ __forceinline__ v16bf frag_b_ldsf32(const float* __restrict__ Bs,
                                                      int ldn, int kk, int n, int lane) {
  const int kbase = (lane < 16) ? 0 : 8;
  const float* p = Bs + (size_t)(kk + kbase) * ldn + n;
  v16bf b;
#pragma unroll
  for (int j = 0; j < 16; ++j) b[j] = (__bf16)p[(size_t)(j < 8 ? j : j + 8) * ldn];
  return b;
}

static __device__ __forceinline__ v8f wmma_bf16(v16bf a, v16bf b, v8f c) {
  return __builtin_amdgcn_wmma_f32_16x16x32_bf16(false, a, false, b, (short)0, c,
                                                 false, false);
}

// ---- Tensor Data Mover: async 2D fp32 tile (rows x cols) -> LDS (contiguous) ----
// D# packing per CDNA5 ISA (group0 128b, group1 256b).
// 6-arg builtin form (amdgpu-toolchain / clang-23): (g0, g1, g2, g3, g4, cpol).
static __device__ __forceinline__ void tdm_load_2d_f32(const float* gsrc,
                                                       unsigned lds_byte_off,
                                                       unsigned rows, unsigned cols,
                                                       unsigned long long row_stride,
                                                       unsigned tensor_rows) {
#if USE_TDM
  const unsigned long long ga = (unsigned long long)(uintptr_t)gsrc;
  u32x4 g0;
  g0[0] = 1u;                                           // count=1, user mode, no gather
  g0[1] = lds_byte_off;                                 // lds_addr [63:32]
  g0[2] = (unsigned)(ga & 0xFFFFFFFFull);               // global_addr [95:64]
  g0[3] = (unsigned)((ga >> 32) & 0x1FFFFFFull) | (2u << 30);  // addr hi + type=2
  const unsigned td0 = (unsigned)row_stride;            // tensor_dim0 (>= cols, in-bounds tiles)
  const unsigned td1 = tensor_rows;                     // tensor_dim1
  i32x8 g1;
  g1[0] = (int)(2u << 16);                              // data_size=2 (4 bytes), mask=0
  g1[1] = (int)((td0 & 0xFFFFu) << 16);                 // tensor_dim0[15:0]
  g1[2] = (int)(((td0 >> 16) & 0xFFFFu) | ((td1 & 0xFFFFu) << 16));
  g1[3] = (int)(((td1 >> 16) & 0xFFFFu) | ((cols & 0xFFFFu) << 16));  // tile_dim0
  g1[4] = (int)(rows & 0xFFFFu);                        // tile_dim1 (tile_dim2=0 -> 2D)
  g1[5] = (int)(row_stride & 0xFFFFFFFFull);            // tensor_dim0_stride lo
  g1[6] = (int)((row_stride >> 32) & 0xFFFFull);        // stride hi, dim1_stride=0
  g1[7] = 0;
  const i32x4 z4 = {0, 0, 0, 0};                        // groups 2/3 unused (2D tensor)
  const i32x8 z8 = {0, 0, 0, 0, 0, 0, 0, 0};            // extra group (unused)
  __builtin_amdgcn_tensor_load_to_lds(g0, g1, z4, z4, z8, 0);
#else
  (void)gsrc; (void)lds_byte_off; (void)rows; (void)cols; (void)row_stride; (void)tensor_rows;
#endif
}

static __device__ __forceinline__ unsigned lds_off(const void* p) {
  return (unsigned)(uintptr_t)p;   // flat LDS address: [31:0] = offset in group segment
}

// ---- Stage 1: routing (one wave32 per token) ----
__global__ void moe1250_routing(const float* __restrict__ x,
                                const float* __restrict__ gw,
                                const float* __restrict__ gb,
                                float* __restrict__ combine) {
  const int lane = threadIdx.x & 31;
  const int tok  = blockIdx.x * (blockDim.x >> 5) + (threadIdx.x >> 5);
  if (tok >= T_TOK) return;
  float acc[NEXP];
#pragma unroll
  for (int e = 0; e < NEXP; ++e) acc[e] = 0.f;
  const float* xr = x + (size_t)tok * HDIM;
  for (int h = lane; h < HDIM; h += 32) {
    const float xv = xr[h];
    const float* g = gw + (size_t)h * NEXP;
#pragma unroll
    for (int e = 0; e < NEXP; ++e) acc[e] += xv * g[e];
  }
#pragma unroll
  for (int e = 0; e < NEXP; ++e) {
#pragma unroll
    for (int off = 16; off > 0; off >>= 1) acc[e] += __shfl_xor(acc[e], off, 32);
  }
  if (lane == 0) {
#pragma unroll
    for (int e = 0; e < NEXP; ++e) acc[e] += gb[e];
    int i0 = 0;  // argmax, ties -> lowest index (matches jax.lax.top_k)
#pragma unroll
    for (int e = 1; e < NEXP; ++e) if (acc[e] > acc[i0]) i0 = e;
    int i1 = (i0 == 0) ? 1 : 0;
#pragma unroll
    for (int e = 0; e < NEXP; ++e) if (e != i0 && acc[e] > acc[i1]) i1 = e;
    const float m  = fmaxf(acc[i0], acc[i1]);
    const float w0 = __expf(acc[i0] - m), w1 = __expf(acc[i1] - m);
    const float inv = 1.f / (w0 + w1);
    float* c = combine + (size_t)tok * NEXP;
#pragma unroll
    for (int e = 0; e < NEXP; ++e) c[e] = 0.f;
    c[i0] = w0 * inv;
    c[i1] = w1 * inv;
  }
}

// ---- Stage 2: per-expert token-list compaction (deterministic token order) ----
__global__ void moe1250_compact(const float* __restrict__ combine,
                                int* __restrict__ offs,
                                int* __restrict__ rowlist) {
  __shared__ int cnt[NEXP];
  __shared__ int base[NEXP + 1];
  const int e = threadIdx.x;
  if (e < NEXP) {
    int c = 0;
    for (int t = 0; t < T_TOK; ++t)
      c += (combine[(size_t)t * NEXP + e] > 0.f) ? 1 : 0;
    cnt[e] = c;
  }
  __syncthreads();
  if (e == 0) {
    int s = 0;
    for (int i = 0; i < NEXP; ++i) { base[i] = s; s += cnt[i]; }
    base[NEXP] = s;  // always == TOPK*T_TOK
    for (int i = 0; i <= NEXP; ++i) offs[i] = base[i];
  }
  __syncthreads();
  if (e < NEXP) {
    int p = base[e];
    for (int t = 0; t < T_TOK; ++t)
      if (combine[(size_t)t * NEXP + e] > 0.f) rowlist[p++] = t;
  }
}

// ---- Unified expert GEMM: block tile 64(M) x 128(N), Kc = 64 ----
// Wave grid 1x8: each wave owns a unique 16-col strip and all 4 M-subtiles
// (acc[4]) -> each B fragment feeds 4 WMMAs, B converted exactly once/block.
// B tiles double-buffered in LDS; TDM loads stage i+1 while computing stage i.
// MODE 0: dst = (x@W1+b1)*(x@W2+b2)   [gathered A rows, dual B]
// MODE 1: dst = silu(A@Wg+bg)
// MODE 2: out += combine * (A@W3+b3)  [scaled scatter, 2 atomics/element total]
template <int MODE, int KDIM, int NDIM>
__global__ void __launch_bounds__(256)
moe1250_gemm(const float* __restrict__ xsrc, const __bf16* __restrict__ asrc,
             const float* __restrict__ WA, const float* __restrict__ WB,
             const float* __restrict__ biasA, const float* __restrict__ biasB,
             const int* __restrict__ offs, const int* __restrict__ rowlist,
             const float* __restrict__ combine,
             __bf16* __restrict__ dst, float* __restrict__ out) {
  const int e    = blockIdx.z;
  const int row0 = offs[e];
  const int cnt  = offs[e + 1] - row0;
  const int tm   = blockIdx.y;
  if (tm * 64 >= cnt) return;
  const int n0 = blockIdx.x * 128;

  __shared__ __bf16 As[64][72];                        // A tile, bf16, +8 pad
  __shared__ float  Bs[2][64][128];                    // B tile ping-pong, fp32 (TDM raw)
  __shared__ float  Bs2[(MODE == 0) ? 2 : 1][64][128]; // second weights (MODE 0)
  __shared__ int    toks[64];

  if (MODE == 0) {
    for (int i = threadIdx.x; i < 64; i += 256) {
      const int row = tm * 64 + i;
      toks[i] = (row < cnt) ? rowlist[row0 + row] : -1;
    }
  }

  const int lane = threadIdx.x & 31;
  const int wave = threadIdx.x >> 5;
  const int nl   = wave * 16 + (lane & 15);   // this wave's column strip

  v8f acc[4] = {};
  v8f acc2[4] = {};                           // MODE 0 only

  const float* WAe = WA + (size_t)e * KDIM * NDIM;
  const float* WBe = (MODE == 0) ? (WB + (size_t)e * KDIM * NDIM) : nullptr;

  constexpr int NST = KDIM / 64;              // K stages
  constexpr int LPS = (MODE == 0) ? 2 : 1;    // TDM loads per stage

#if USE_TDM
  if (wave == 0) {                            // prologue: stage 0 DMA
    tdm_load_2d_f32(WAe + n0, lds_off(&Bs[0][0][0]),
                    64u, 128u, (unsigned long long)NDIM, (unsigned)KDIM);
    if (MODE == 0)
      tdm_load_2d_f32(WBe + n0, lds_off(&Bs2[0][0][0]),
                      64u, 128u, (unsigned long long)NDIM, (unsigned)KDIM);
  }
#endif

  for (int i = 0; i < NST; ++i) {
    const int kb = i * 64;
    __syncthreads();  // compute(i-1) done: safe to overwrite As and buffer (i+1)&1

    // --- stage A tile (all waves; overlaps with in-flight TDM) ---
    if (MODE == 0) {
      for (int t = threadIdx.x; t < 64 * 16; t += 256) {  // 64 rows x 64 cols / 4
        const int r = t >> 4, c4 = t & 15;
        float4 v = make_float4(0.f, 0.f, 0.f, 0.f);
        const int tok = toks[r];
        if (tok >= 0)
          v = *(const float4*)(xsrc + (size_t)tok * KDIM + kb + (size_t)c4 * 4);
        __bf16* d = &As[r][c4 * 4];
        d[0] = (__bf16)v.x; d[1] = (__bf16)v.y; d[2] = (__bf16)v.z; d[3] = (__bf16)v.w;
      }
    } else {
      for (int t = threadIdx.x; t < 64 * 8; t += 256) {   // 64 rows x 64 cols / 8
        const int r = t >> 3, c8 = t & 7;
        uint4 v = make_uint4(0, 0, 0, 0);
        const int row = tm * 64 + r;
        if (row < cnt)
          v = *(const uint4*)(asrc + (size_t)(row0 + row) * KDIM + kb + (size_t)c8 * 8);
        *(uint4*)&As[r][c8 * 8] = v;
      }
    }

#if USE_TDM
    if (wave == 0) {
      if (i + 1 < NST) {                      // prefetch next stage via TDM
        tdm_load_2d_f32(WAe + (size_t)(kb + 64) * NDIM + n0,
                        lds_off(&Bs[(i + 1) & 1][0][0]),
                        64u, 128u, (unsigned long long)NDIM, (unsigned)KDIM);
        if (MODE == 0)
          tdm_load_2d_f32(WBe + (size_t)(kb + 64) * NDIM + n0,
                          lds_off(&Bs2[(i + 1) & 1][0][0]),
                          64u, 128u, (unsigned long long)NDIM, (unsigned)KDIM);
        // TDM completes in-order: <=LPS outstanding => stage i has landed.
        __builtin_amdgcn_s_wait_tensorcnt((short)LPS);
      } else {
        __builtin_amdgcn_s_wait_tensorcnt((short)0);
      }
    }
#else
    for (int t = threadIdx.x; t < 64 * 32; t += 256) {    // 64 rows x 128 cols / 4
      const int r = t >> 5, c4 = t & 31;
      *(float4*)&Bs[i & 1][r][c4 * 4] =
          *(const float4*)(WAe + (size_t)(kb + r) * NDIM + n0 + c4 * 4);
    }
    if (MODE == 0) {
      for (int t = threadIdx.x; t < 64 * 32; t += 256) {
        const int r = t >> 5, c4 = t & 31;
        *(float4*)&Bs2[i & 1][r][c4 * 4] =
            *(const float4*)(WBe + (size_t)(kb + r) * NDIM + n0 + c4 * 4);
      }
    }
#endif
    __syncthreads();

    // --- compute: per kk one B fragment feeds 4 (M-subtile) WMMAs ---
    const float* Bcur  = &Bs[i & 1][0][0];
    const float* B2cur = (MODE == 0) ? &Bs2[i & 1][0][0] : nullptr;
#pragma unroll
    for (int kk = 0; kk < 64; kk += 32) {
      const v16bf b = frag_b_ldsf32(Bcur, 128, kk, nl, lane);
      v16bf b2 = b;
      if (MODE == 0) b2 = frag_b_ldsf32(B2cur, 128, kk, nl, lane);
#pragma unroll
      for (int mi = 0; mi < 4; ++mi) {
        const v16bf a = frag_a_lds(&As[mi * 16 + (lane & 15)][0], kk, lane);
        acc[mi] = wmma_bf16(a, b, acc[mi]);
        if (MODE == 0) acc2[mi] = wmma_bf16(a, b2, acc2[mi]);
      }
    }
  }

  // --- epilogue (C/D layout: vgpr r -> row r, +8 for lanes 16-31) ----
  const int ng = n0 + nl;
  const int mo = (lane < 16) ? 0 : 8;
  const float bA = biasA[(size_t)e * NDIM + ng];
  const float bB = (MODE == 0) ? biasB[(size_t)e * NDIM + ng] : 0.f;
#pragma unroll
  for (int mi = 0; mi < 4; ++mi) {
#pragma unroll
    for (int r = 0; r < 8; ++r) {
      const int row = tm * 64 + mi * 16 + r + mo;
      if (row < cnt) {
        if (MODE == 0) {
          dst[(size_t)(row0 + row) * NDIM + ng] =
              (__bf16)((acc[mi][r] + bA) * (acc2[mi][r] + bB));
        } else if (MODE == 1) {
          const float v = acc[mi][r] + bA;
          dst[(size_t)(row0 + row) * NDIM + ng] = (__bf16)(v / (1.f + __expf(-v)));
        } else {
          const int tok = rowlist[row0 + row];
          const float w = combine[(size_t)tok * NEXP + e];
          unsafeAtomicAdd(out + (size_t)tok * NDIM + ng, (acc[mi][r] + bA) * w);
        }
      }
    }
  }
}

extern "C" void kernel_launch(void* const* d_in, const int* in_sizes, int n_in,
                              void* d_out, int out_size, void* d_ws, size_t ws_size,
                              hipStream_t stream) {
  const float* x  = (const float*)d_in[0];
  const float* W1 = (const float*)d_in[1];
  const float* b1 = (const float*)d_in[2];
  const float* W2 = (const float*)d_in[3];
  const float* b2 = (const float*)d_in[4];
  const float* Wg = (const float*)d_in[5];
  const float* bg = (const float*)d_in[6];
  const float* W3 = (const float*)d_in[7];
  const float* b3 = (const float*)d_in[8];
  const float* gw = (const float*)d_in[9];
  const float* gb = (const float*)d_in[10];
  (void)in_sizes; (void)n_in; (void)out_size; (void)ws_size;

  // Workspace carve-up (~32.2 MB total)
  char* w = (char*)d_ws;
  float* combine = (float*)w;  w += (size_t)T_TOK * NEXP * sizeof(float);
  int*   offs    = (int*)w;    w += 256;
  int*   rowlist = (int*)w;    w += (size_t)TOPK * T_TOK * sizeof(int);
  w = (char*)(((uintptr_t)w + 255) & ~(uintptr_t)255);
  __bf16* hm   = (__bf16*)w;   w += (size_t)TOPK * T_TOK * FDIM * sizeof(__bf16);
  __bf16* gbuf = (__bf16*)w;   w += (size_t)TOPK * T_TOK * FDIM * sizeof(__bf16);

  (void)hipMemsetAsync(d_out, 0, (size_t)T_TOK * HDIM * sizeof(float), stream);

  moe1250_routing<<<dim3(T_TOK / 8), dim3(256), 0, stream>>>(x, gw, gb, combine);
  moe1250_compact<<<dim3(1), dim3(32), 0, stream>>>(combine, offs, rowlist);

  moe1250_gemm<0, HDIM, FDIM><<<dim3(FDIM / 128, T_TOK / 64, NEXP), dim3(256), 0, stream>>>(
      x, nullptr, W1, W2, b1, b2, offs, rowlist, nullptr, hm, nullptr);
  moe1250_gemm<1, FDIM, FDIM><<<dim3(FDIM / 128, T_TOK / 64, NEXP), dim3(256), 0, stream>>>(
      nullptr, hm, Wg, nullptr, bg, nullptr, offs, rowlist, nullptr, gbuf, nullptr);
  moe1250_gemm<2, FDIM, HDIM><<<dim3(HDIM / 128, T_TOK / 64, NEXP), dim3(256), 0, stream>>>(
      nullptr, gbuf, W3, nullptr, b3, nullptr, offs, rowlist, combine, nullptr, (float*)d_out);
}
